// MultiHeadCausalAttention_74208444940647
// MI455X (gfx1250) — matrix operand order repaired and checked by reading, not compiled, not run on hardware
//
#include <hip/hip_runtime.h>
#include <hip/hip_bf16.h>

// ---------------------------------------------------------------------------
// MultiHeadCausalAttention for MI455X (gfx1250, wave32, WMMA bf16)
// B=4, T=2048, D=1024, H=16, HD=64.  Matmuls via v_wmma_f32_16x16x32_bf16.
// Data movement: global_load_async_to_lds / global_store_async_from_lds
// (ASYNCcnt) + TDM tensor_load_to_lds (TENSORcnt, 6-arg form).
// Weights pre-transposed once so all hot-loop staging is contiguous async.
// Confirmed available (unused for now, per-lane layout undocumented):
//   __builtin_amdgcn_ds_load_tr16_b128_v8i16/_v8bf16,
//   __builtin_amdgcn_global_load_tr16_b128_v8i16,
//   __builtin_amdgcn_global_load_tr8_b64_v2i32,
//   __builtin_amdgcn_ds_atomic_async_barrier_arrive_b64
// ---------------------------------------------------------------------------

typedef unsigned short u16;
typedef __attribute__((ext_vector_type(16))) __bf16        v16bf;
typedef __attribute__((ext_vector_type(8)))  float         v8f;
typedef __attribute__((ext_vector_type(4)))  int           v4i_t;
typedef __attribute__((ext_vector_type(8)))  int           v8i_t;
typedef __attribute__((ext_vector_type(4)))  unsigned int  v4u_t;

#define AS1 __attribute__((address_space(1)))
#define AS3 __attribute__((address_space(3)))

#if __has_builtin(__builtin_amdgcn_global_load_async_to_lds_b128)
#define HAVE_ASYNC 1
#endif
#if __has_builtin(__builtin_amdgcn_global_store_async_from_lds_b128)
#define HAVE_ASYNC_ST 1
#endif
#if __has_builtin(__builtin_amdgcn_s_wait_asynccnt)
#define HAVE_WAIT_ASYNC 1
#endif
#if __has_builtin(__builtin_amdgcn_tensor_load_to_lds)
#define HAVE_TDM 1
#endif

union FragBF {
    uint4 q[2];   // 32 bytes = 16 bf16
    v16bf v;
};

__device__ __forceinline__ u16 f2bf(float f) {
    unsigned int u = __float_as_uint(f);
    unsigned int r = (u + 0x7FFFu + ((u >> 16) & 1u)) >> 16;   // RNE
    return (u16)r;
}

// async 16B global -> LDS copy (per-lane addresses); params are non-const
// v4i pointers in AS1/AS3 (probe-confirmed signature).
__device__ __forceinline__ void cp_async_b128(void* lds, const void* g) {
#if defined(HAVE_ASYNC)
    __builtin_amdgcn_global_load_async_to_lds_b128(
        (AS1 v4i_t*)g, (AS3 v4i_t*)lds, 0, 0);
#else
    *(uint4*)lds = *(const uint4*)g;
#endif
}

// async 16B LDS -> global copy (per-lane addresses)
__device__ __forceinline__ void st_async_b128(void* g, const void* lds) {
#if defined(HAVE_ASYNC_ST)
    __builtin_amdgcn_global_store_async_from_lds_b128(
        (AS1 v4i_t*)g, (AS3 v4i_t*)lds, 0, 0);
#else
    *(uint4*)g = *(const uint4*)lds;
#endif
}

__device__ __forceinline__ void wait_async0() {
#if defined(HAVE_WAIT_ASYNC)
    __builtin_amdgcn_s_wait_asynccnt(0);
#else
    asm volatile("s_wait_asynccnt 0x0" ::: "memory");
#endif
}

#if defined(HAVE_TDM)
// TDM: load one 64x64 bf16 tile (row-major, tensor row stride 64 elems) from
// global into LDS with 8-u16 row padding (pad_interval=32 dwords -> one row,
// pad_amount=4 dwords) so the LDS row stride is 72 u16 = 144 B.
__device__ __forceinline__ void tdm_load_tile64x64(unsigned lds_byte_addr,
                                                   const void* gsrc) {
    unsigned long long ga = (unsigned long long)gsrc;
    v4u_t g0;
    g0[0] = 1u;                                          // count=1, user mode
    g0[1] = lds_byte_addr;                               // lds_addr
    g0[2] = (unsigned)(ga & 0xffffffffu);                // global_addr[31:0]
    g0[3] = (unsigned)((ga >> 32) & 0x01ffffffu)         // global_addr[56:32]
          | (2u << 30);                                  // type = 2 (image)
    v8i_t g1;
    g1[0] = (1 << 16)        // data_size = 1 -> 2 bytes
          | (1 << 20)        // pad_enable
          | (4 << 22)        // pad_interval: 4 -> 32 dwords (= one 64-elem row)
          | (3 << 25);       // pad_amount:   3 -> 4 dwords (= 8 u16)
    g1[1] = (64 << 16);      // tensor_dim0 = 64 (bits 79:48, low half)
    g1[2] = (2048 << 16);    // tensor_dim0 hi = 0 | tensor_dim1 = 2048 (lo)
    g1[3] = (64 << 16);      // tensor_dim1 hi = 0 | tile_dim0 = 64
    g1[4] = 64;              // tile_dim1 = 64, tile_dim2 = 0
    g1[5] = 64;              // tensor_dim0_stride = 64 (bits 207:160, low)
    g1[6] = 0;               // stride hi | tensor_dim1_stride lo
    g1[7] = 0;
    v4i_t z4 = {0, 0, 0, 0};
    v8i_t z8 = {0, 0, 0, 0, 0, 0, 0, 0};
    __builtin_amdgcn_tensor_load_to_lds(g0, g1, z4, z4, z8, 0);
    __builtin_amdgcn_s_wait_tensorcnt((short)0);
}
#endif

// ---------------------------------------------------------------------------
// fp32 -> bf16 conversion (4 elements/thread)
// ---------------------------------------------------------------------------
__global__ __launch_bounds__(256)
void mhca_convert_bf16(const float* __restrict__ src, u16* __restrict__ dst, int n) {
    int i = (blockIdx.x * 256 + threadIdx.x) * 4;
    if (i + 3 < n) {
        float4 v = *(const float4*)(src + i);
        ushort4 o;
        o.x = f2bf(v.x); o.y = f2bf(v.y); o.z = f2bf(v.z); o.w = f2bf(v.w);
        *(ushort4*)(dst + i) = o;
    }
}

// ---------------------------------------------------------------------------
// fp32 [rows][cols] -> bf16 transposed [cols][rows] (32x32 LDS-tiled)
// grid (rows/32, cols/32)
// ---------------------------------------------------------------------------
__global__ __launch_bounds__(256)
void mhca_convert_bf16_T(const float* __restrict__ src, u16* __restrict__ dst,
                         int rows, int cols) {
    __shared__ u16 tile[32][33];
    const int k0 = blockIdx.x * 32;
    const int n0 = blockIdx.y * 32;
    const int r = threadIdx.x >> 5;
    const int c = threadIdx.x & 31;
#pragma unroll
    for (int j = 0; j < 4; ++j) {
        int k = r + j * 8;
        tile[k][c] = f2bf(src[(size_t)(k0 + k) * cols + n0 + c]);
    }
    __syncthreads();
#pragma unroll
    for (int j = 0; j < 4; ++j) {
        int n = r + j * 8;
        dst[(size_t)(n0 + n) * rows + k0 + c] = tile[c][n];
    }
}

// ---------------------------------------------------------------------------
// Shared GEMM mainloop: 128x128 block tile, K-step 32, 8 waves (4Mx2N),
// wave tile 32x64 = 2x4 WMMA accumulators.
// As: [row][k], Bs: [n][k] -- both k-contiguous in global (Wt pre-transposed),
// both staged with async global->LDS.  Row stride 40 u16 = 80 B (16B aligned).
// ---------------------------------------------------------------------------
#define LDK 40

__device__ __forceinline__ void gemm_mainloop_bf16(
    const u16* __restrict__ A,  int lda,    // A  [M][Kdim]
    const u16* __restrict__ Wt, int ldw,    // Wt [N][Kdim]
    int rowBase, int colBase, int Kdim,
    u16 As[][LDK], u16 Bs[][LDK], v8f acc[2][4])
{
    const int tid   = threadIdx.x;
    const int lane  = tid & 31;
    const int wid   = tid >> 5;
    const int wm    = (wid & 3) * 32;
    const int wn    = (wid >> 2) * 64;
    const int lhalf = lane >> 4;
    const int l15   = lane & 15;

    for (int k0 = 0; k0 < Kdim; k0 += 32) {
        // --- stage A tile 128x32 and B tile 128x32, all async (4x16B/thread)
        {
            int r = tid >> 1, h = (tid & 1) * 16;
            const u16* sa = A  + (size_t)(rowBase + r) * lda + k0 + h;
            const u16* sb = Wt + (size_t)(colBase + r) * ldw + k0 + h;
            cp_async_b128(&As[r][h],     sa);
            cp_async_b128(&As[r][h + 8], sa + 8);
            cp_async_b128(&Bs[r][h],     sb);
            cp_async_b128(&Bs[r][h + 8], sb + 8);
        }
        wait_async0();
        __syncthreads();

        // --- fragments ---
        FragBF a[2], b[4];
#pragma unroll
        for (int tm = 0; tm < 2; ++tm) {
            int rb  = wm + tm * 16 + l15;
            int klo = lhalf * 8;                    // ISA A 16x32 interleave
            a[tm].q[0] = *(const uint4*)&As[rb][klo];
            a[tm].q[1] = *(const uint4*)&As[rb][klo + 16];
        }
#pragma unroll
        for (int tn = 0; tn < 4; ++tn) {
            int cb = wn + tn * 16 + l15;
            int kb = lhalf * 8;                     // ISA B 32x16: same K interleave as A
            b[tn].q[0] = *(const uint4*)&Bs[cb][kb];
            b[tn].q[1] = *(const uint4*)&Bs[cb][kb + 16];
        }
#pragma unroll
        for (int tm = 0; tm < 2; ++tm)
#pragma unroll
            for (int tn = 0; tn < 4; ++tn)
                acc[tm][tn] = __builtin_amdgcn_wmma_f32_16x16x32_bf16(
                    false, a[tm].v, false, b[tn].v, (short)0, acc[tm][tn],
                    false, false);
        __syncthreads();
    }
}

// ---------------------------------------------------------------------------
// QKV GEMM: [8192,1024] x Wt[3072,1024] -> Q/K as [B,H,T,HD], V as [B,H,HD,T]
// grid (64, 24)
// ---------------------------------------------------------------------------
__global__ __launch_bounds__(256)
void mhca_qkv_gemm(const u16* __restrict__ A, const u16* __restrict__ Wt,
                   u16* __restrict__ Qo, u16* __restrict__ Ko, u16* __restrict__ Vo)
{
    __shared__ u16 As[128][LDK];
    __shared__ u16 Bs[128][LDK];
    v8f acc[2][4] = {};
    const int rowBase = blockIdx.x * 128;
    const int colBase = blockIdx.y * 128;
    gemm_mainloop_bf16(A, 1024, Wt, 1024, rowBase, colBase, 1024, As, Bs, acc);

    const int lane = threadIdx.x & 31;
    const int wid  = threadIdx.x >> 5;
    const int wm   = (wid & 3) * 32;
    const int wn   = (wid >> 2) * 64;
    const int hi   = (lane >> 4) * 8;
    const int l15  = lane & 15;
    const int dbase = colBase & 1023;

    if (colBase < 2048) {
        // Q / K output in [B,H,T,HD]
        u16* dst = (colBase < 1024) ? Qo : Ko;
#pragma unroll
        for (int tm = 0; tm < 2; ++tm)
#pragma unroll
            for (int tn = 0; tn < 4; ++tn)
#pragma unroll
                for (int r = 0; r < 8; ++r) {
                    int row = rowBase + wm + tm * 16 + r + hi;
                    int col = dbase + wn + tn * 16 + l15;
                    int b = row >> 11, t = row & 2047;
                    int h = col >> 6,  hd = col & 63;
                    size_t idx = ((((size_t)b * 16 + h) * 2048) + t) * 64 + hd;
                    dst[idx] = f2bf(acc[tm][tn][r]);
                }
    } else {
        // V output transposed: [B,H,HD,T]
#pragma unroll
        for (int tm = 0; tm < 2; ++tm)
#pragma unroll
            for (int tn = 0; tn < 4; ++tn)
#pragma unroll
                for (int r = 0; r < 8; ++r) {
                    int row = rowBase + wm + tm * 16 + r + hi;
                    int col = dbase + wn + tn * 16 + l15;
                    int b = row >> 11, t = row & 2047;
                    int h = col >> 6,  hd = col & 63;
                    size_t idx = ((((size_t)b * 16 + h) * 64) + hd) * 2048 + t;
                    Vo[idx] = f2bf(acc[tm][tn][r]);
                }
    }
}

// ---------------------------------------------------------------------------
// Flash attention (causal): one block per (b,h, 128-query tile), 8 waves.
// Q via async-LDS copies; K via TDM DMA (wave 0); V (pre-transposed in global)
// via async copies; output staged in LDS, written with async b128 stores.
// grid (16, 64)
// ---------------------------------------------------------------------------
__global__ __launch_bounds__(256)
void mhca_flash_attn(const u16* __restrict__ Q, const u16* __restrict__ K,
                     const u16* __restrict__ V, u16* __restrict__ O)
{
    __shared__ u16 Qs[128][72];    // [qrow][hd]   stride 144B
    __shared__ u16 Ks[64][72];     // [key][hd]    (TDM pads rows to 72)
    __shared__ u16 Vt[64][72];     // [hd][key]
    __shared__ u16 Ps[8][16][72];  // per-wave P / O-staging tile

    const int qb = blockIdx.x;
    const int bh = blockIdx.y;
    const int qBase = qb * 128;
    const u16* Qp = Q + (size_t)bh * 2048 * 64;   // [t][hd]
    const u16* Kp = K + (size_t)bh * 2048 * 64;   // [t][hd]
    const u16* Vp = V + (size_t)bh * 64 * 2048;   // [hd][t]  (pre-transposed)

    const int tid = threadIdx.x, lane = tid & 31, wid = tid >> 5;
    const int l15 = lane & 15, lhalf = lane >> 4;
    const int qrow0 = wid * 16;

    // stage Q tile via async copies (4x16B per thread)
    {
        int r = tid >> 1, h = (tid & 1) * 32;
        const u16* src = Qp + (size_t)(qBase + r) * 64 + h;
        cp_async_b128(&Qs[r][h],      src);
        cp_async_b128(&Qs[r][h + 8],  src + 8);
        cp_async_b128(&Qs[r][h + 16], src + 16);
        cp_async_b128(&Qs[r][h + 24], src + 24);
    }

    v8f accO[4] = {};
    float mrow[8], lrow[8];
#pragma unroll
    for (int r = 0; r < 8; ++r) { mrow[r] = -1e30f; lrow[r] = 0.0f; }

    const int nKB = (qBase >> 6) + 2;   // causal: key blocks covering [0, qBase+128)
    for (int kb = 0; kb < nKB; ++kb) {
        const int kBase = kb * 64;
        wait_async0();
        __syncthreads();   // prev-iter LDS consumers done (and Q tile visible)

        // ---- stage K tile: TDM DMA (wave 0) or per-thread async copies ----
#if defined(HAVE_TDM)
        if (wid == 0) {
            tdm_load_tile64x64((unsigned)(size_t)&Ks[0][0],
                               Kp + (size_t)kBase * 64);
        }
#else
        {
            int kr = tid >> 2, c = (tid & 3) * 16;
            const u16* sk = Kp + (size_t)(kBase + kr) * 64 + c;
            cp_async_b128(&Ks[kr][c],     sk);
            cp_async_b128(&Ks[kr][c + 8], sk + 8);
        }
#endif
        // ---- stage V tile (already [hd][t] in global): contiguous async ----
        {
            int hd = tid >> 2, c = (tid & 3) * 16;
            const u16* sv = Vp + (size_t)hd * 2048 + kBase + c;
            cp_async_b128(&Vt[hd][c],     sv);
            cp_async_b128(&Vt[hd][c + 8], sv + 8);
        }
        if (kb + 1 < nKB) {   // global_prefetch hint for next key block
            __builtin_prefetch(Kp + (size_t)(kBase + 64) * 64, 0, 1);
            __builtin_prefetch(Vp + (size_t)(kBase + 64), 0, 1);
        }
        wait_async0();
        __syncthreads();

        // ---- S = (Q Kt) * 1/sqrt(64) ----
        FragBF aQ[2];
#pragma unroll
        for (int s = 0; s < 2; ++s) {
            int rb  = qrow0 + l15;
            int klo = s * 32 + lhalf * 8;
            aQ[s].q[0] = *(const uint4*)&Qs[rb][klo];
            aQ[s].q[1] = *(const uint4*)&Qs[rb][klo + 16];
        }
        v8f S[4] = {};
#pragma unroll
        for (int n = 0; n < 4; ++n)
#pragma unroll
            for (int s = 0; s < 2; ++s) {
                FragBF bK;
                int cb = n * 16 + l15;
                int kk = s * 32 + lhalf * 8;
                bK.q[0] = *(const uint4*)&Ks[cb][kk];
                bK.q[1] = *(const uint4*)&Ks[cb][kk + 16];
                S[n] = __builtin_amdgcn_wmma_f32_16x16x32_bf16(
                    false, aQ[s].v, false, bK.v, (short)0, S[n], false, false);
            }

        // ---- scale + causal mask + row max ----
        float rmax[8];
#pragma unroll
        for (int r = 0; r < 8; ++r) rmax[r] = -1e30f;
#pragma unroll
        for (int n = 0; n < 4; ++n)
#pragma unroll
            for (int r = 0; r < 8; ++r) {
                int qi = qBase + qrow0 + r + lhalf * 8;
                int ki = kBase + n * 16 + l15;
                float sv = S[n][r] * 0.125f;
                sv = (ki <= qi) ? sv : -1.0e10f;
                S[n][r] = sv;
                rmax[r] = fmaxf(rmax[r], sv);
            }
#pragma unroll
        for (int r = 0; r < 8; ++r) {
            float v = rmax[r];
            v = fmaxf(v, __shfl_xor(v, 1, 32));
            v = fmaxf(v, __shfl_xor(v, 2, 32));
            v = fmaxf(v, __shfl_xor(v, 4, 32));
            v = fmaxf(v, __shfl_xor(v, 8, 32));
            rmax[r] = v;
        }
        float corr[8];
#pragma unroll
        for (int r = 0; r < 8; ++r) {
            float mnew = fmaxf(mrow[r], rmax[r]);
            corr[r] = __expf(mrow[r] - mnew);
            mrow[r] = mnew;
        }

        // ---- P = exp(S - m), row sums, stash P (bf16) in per-wave LDS ----
        float rsum[8];
#pragma unroll
        for (int r = 0; r < 8; ++r) rsum[r] = 0.0f;
#pragma unroll
        for (int n = 0; n < 4; ++n)
#pragma unroll
            for (int r = 0; r < 8; ++r) {
                float p = __expf(S[n][r] - mrow[r]);
                rsum[r] += p;
                Ps[wid][r + lhalf * 8][n * 16 + l15] = f2bf(p);
            }
#pragma unroll
        for (int r = 0; r < 8; ++r) {
            float v = rsum[r];
            v += __shfl_xor(v, 1, 32);
            v += __shfl_xor(v, 2, 32);
            v += __shfl_xor(v, 4, 32);
            v += __shfl_xor(v, 8, 32);
            lrow[r] = lrow[r] * corr[r] + v;
        }
        // rescale running output
#pragma unroll
        for (int t = 0; t < 4; ++t)
#pragma unroll
            for (int r = 0; r < 8; ++r) accO[t][r] *= corr[r];

        // P writes are same-wave LDS; wait before reading fragments back
        asm volatile("s_wait_dscnt 0x0" ::: "memory");

        // ---- O += P @ V ----
#pragma unroll
        for (int hdt = 0; hdt < 4; ++hdt)
#pragma unroll
            for (int s = 0; s < 2; ++s) {
                FragBF aP, bV;
                int klo = s * 32 + lhalf * 8;
                aP.q[0] = *(const uint4*)&Ps[wid][l15][klo];
                aP.q[1] = *(const uint4*)&Ps[wid][l15][klo + 16];
                int cb = hdt * 16 + l15;
                int kk = s * 32 + lhalf * 8;
                bV.q[0] = *(const uint4*)&Vt[cb][kk];
                bV.q[1] = *(const uint4*)&Vt[cb][kk + 16];
                accO[hdt] = __builtin_amdgcn_wmma_f32_16x16x32_bf16(
                    false, aP.v, false, bV.v, (short)0, accO[hdt], false, false);
            }
    }

    // ---- epilogue: O /= l -> stage bf16 in per-wave LDS -> async b128 stores
    const int h = bh & 15, b = bh >> 4;
#pragma unroll
    for (int hdt = 0; hdt < 4; ++hdt)
#pragma unroll
        for (int r = 0; r < 8; ++r) {
            float val = accO[hdt][r] / lrow[r];
            Ps[wid][r + lhalf * 8][hdt * 16 + l15] = f2bf(val);
        }
    asm volatile("s_wait_dscnt 0x0" ::: "memory");   // LDS staged before async read
#pragma unroll
    for (int j = 0; j < 4; ++j) {
        int chunk = lane + 32 * j;          // 0..127 (16 rows x 8 chunks)
        int rt    = chunk >> 3;             // row within wave tile
        int off   = (chunk & 7) * 8;        // u16 offset within row
        int qi    = qBase + qrow0 + rt;
        u16* gdst = O + ((size_t)(b * 2048 + qi)) * 1024 + h * 64 + off;
        st_async_b128(gdst, &Ps[wid][rt][off]);
    }
    wait_async0();
}

// ---------------------------------------------------------------------------
// Output projection: attnOut[8192,1024] x Wt[1024,1024] + bias + residual
// grid (64, 8)
// ---------------------------------------------------------------------------
__global__ __launch_bounds__(256)
void mhca_proj_gemm(const u16* __restrict__ A, const u16* __restrict__ Wt,
                    const float* __restrict__ bias, const float* __restrict__ resid,
                    float* __restrict__ X)
{
    __shared__ u16 As[128][LDK];
    __shared__ u16 Bs[128][LDK];
    v8f acc[2][4] = {};
    const int rowBase = blockIdx.x * 128;
    const int colBase = blockIdx.y * 128;
    gemm_mainloop_bf16(A, 1024, Wt, 1024, rowBase, colBase, 1024, As, Bs, acc);

    const int lane = threadIdx.x & 31;
    const int wid  = threadIdx.x >> 5;
    const int wm   = (wid & 3) * 32;
    const int wn   = (wid >> 2) * 64;
    const int hi   = (lane >> 4) * 8;
    const int l15  = lane & 15;

#pragma unroll
    for (int tm = 0; tm < 2; ++tm)
#pragma unroll
        for (int tn = 0; tn < 4; ++tn)
#pragma unroll
            for (int r = 0; r < 8; ++r) {
                int row = rowBase + wm + tm * 16 + r + hi;
                int col = colBase + wn + tn * 16 + l15;
                size_t idx = (size_t)row * 1024 + col;
                X[idx] = acc[tm][tn][r] + bias[col] + resid[idx];
            }
}

// ---------------------------------------------------------------------------
// LayerNorm: one wave32 per 1024-element row, 8 rows per block. grid (1024)
// ---------------------------------------------------------------------------
__global__ __launch_bounds__(256)
void mhca_layernorm(const float* __restrict__ X, const float* __restrict__ gamma,
                    const float* __restrict__ beta, float* __restrict__ out)
{
    const int row  = blockIdx.x * 8 + (threadIdx.x >> 5);
    const int lane = threadIdx.x & 31;
    const float* x = X + (size_t)row * 1024;

    float s = 0.0f, s2 = 0.0f;
    float vals[32];
#pragma unroll
    for (int i = 0; i < 32; ++i) {
        float v = x[lane + i * 32];
        vals[i] = v;
        s += v;
        s2 += v * v;
    }
#pragma unroll
    for (int m = 16; m >= 1; m >>= 1) {
        s  += __shfl_xor(s,  m, 32);
        s2 += __shfl_xor(s2, m, 32);
    }
    const float mu  = s * (1.0f / 1024.0f);
    const float var = s2 * (1.0f / 1024.0f) - mu * mu;
    const float inv = rsqrtf(var + 1e-5f);

    float* o = out + (size_t)row * 1024;
#pragma unroll
    for (int i = 0; i < 32; ++i) {
        int c = lane + i * 32;
        o[c] = (vals[i] - mu) * inv * gamma[c] + beta[c];
    }
}

// ---------------------------------------------------------------------------
// Host-side orchestration
// ---------------------------------------------------------------------------
extern "C" void kernel_launch(void* const* d_in, const int* in_sizes, int n_in,
                              void* d_out, int out_size, void* d_ws, size_t ws_size,
                              hipStream_t stream)
{
    const float* target  = (const float*)d_in[0];   // [4,2048,1024]
    const float* w_qkv   = (const float*)d_in[1];   // [1024,3072]
    const float* w_proj  = (const float*)d_in[2];   // [1024,1024]
    const float* b_proj  = (const float*)d_in[3];   // [1024]
    const float* ln_gamma= (const float*)d_in[4];   // [1024]
    const float* ln_beta = (const float*)d_in[5];   // [1024]
    float* out = (float*)d_out;

    const size_t MD = (size_t)8192 * 1024;          // B*T*D elements

    u16* aBF     = (u16*)d_ws;                      // target bf16 [8192,1024]
    u16* wqkvT   = aBF + MD;                        // W_qkv^T bf16 [3072,1024]
    u16* wprojT  = wqkvT + (size_t)1024 * 3072;     // W_proj^T bf16 [1024,1024]
    u16* Qb      = wprojT + (size_t)1024 * 1024;    // [B,H,T,HD]
    u16* Kb      = Qb + MD;                         // [B,H,T,HD]
    u16* Vb      = Kb + MD;                         // [B,H,HD,T] (transposed)
    u16* attnO   = Vb + MD;                         // [8192,1024] bf16
    float* Xb    = (float*)(attnO + MD);            // pre-LN fp32 [8192,1024]

    // 1) fp32 -> bf16 (activations plain; weights transposed once)
    mhca_convert_bf16<<<8192, 256, 0, stream>>>(target, aBF, (int)MD);
    mhca_convert_bf16_T<<<dim3(32, 96), 256, 0, stream>>>(w_qkv,  wqkvT,  1024, 3072);
    mhca_convert_bf16_T<<<dim3(32, 32), 256, 0, stream>>>(w_proj, wprojT, 1024, 1024);

    // 2) QKV projection -> Q/K [B,H,T,HD], V [B,H,HD,T]
    mhca_qkv_gemm<<<dim3(64, 24), 256, 0, stream>>>(aBF, wqkvT, Qb, Kb, Vb);

    // 3) causal flash attention -> attnO [8192,1024] bf16
    mhca_flash_attn<<<dim3(16, 64), 256, 0, stream>>>(Qb, Kb, Vb, attnO);

    // 4) output projection + bias + residual -> Xb fp32
    mhca_proj_gemm<<<dim3(64, 8), 256, 0, stream>>>(attnO, wprojT, b_proj, target, Xb);

    // 5) LayerNorm -> d_out
    mhca_layernorm<<<1024, 256, 0, stream>>>(Xb, ln_gamma, ln_beta, out);
}
